// FHEAttention_7679401525929
// MI455X (gfx1250) — matrix operand order repaired and checked by reading, not compiled, not run on hardware
//
#include <hip/hip_runtime.h>
#include <hip/hip_bf16.h>
#include <stdint.h>

// ---------------------------------------------------------------------------
// CDNA5 (gfx1250) bf16 WMMA implementation of polynomial-softmax attention.
//   q = (x Wq^T)*0.1 ; k = (x Wk^T)*0.1 ; v = x Wv^T
//   s = 0.1 * q k^T ; s = s*s + s ; ao = s v ; out = (ao Wo^T)*0.1
// GEMMs: v_wmma_f32_16x16x32_bf16, 128x128 block tile, 8 waves x (64x32),
// double-buffered LDS fed by global_load_async_to_lds_b128 (ASYNCcnt path).
// ---------------------------------------------------------------------------

typedef __attribute__((ext_vector_type(16))) __bf16    v16bf;
typedef __attribute__((ext_vector_type(8)))  float     v8f;

static __device__ __forceinline__ uint16_t f2bf(float f) {
  union { float f; uint32_t u; } c; c.f = f;
  uint32_t u = c.u;
  uint32_t r = (u + 0x7FFFu + ((u >> 16) & 1u)) >> 16;  // round-to-nearest-even
  return (uint16_t)r;
}

// Per-lane async copy of 16 bytes global -> LDS, tracked by ASYNCcnt.
static __device__ __forceinline__ void async_copy_b128(uint32_t lds_off,
                                                       const void* gptr) {
  asm volatile("global_load_async_to_lds_b128 %0, %1, off"
               :: "v"(lds_off), "v"(gptr)
               : "memory");
}
static __device__ __forceinline__ void wait_asynccnt0() {
  asm volatile("s_wait_asynccnt 0x0" ::: "memory");
}

enum { EPI_BF16 = 0, EPI_SCORES = 1, EPI_BF16T = 2, EPI_F32 = 3 };

// C(MxN) = A(MxK, row-major) * B(KxN); Bm holds B column-major, i.e. Bm is
// NxK row-major (row n = column n of B, contiguous over K).
// Block tile 128x128, 256 threads = 8 wave32s in a 2x4 grid, wave tile 64x32.
template<int EPI>
__global__ __launch_bounds__(256)
void gemm_wmma_bf16(const uint16_t* __restrict__ A,
                    const uint16_t* __restrict__ Bm,
                    void* __restrict__ Cp,
                    int K, int lda, int ldb, int ldc,
                    long sA, long sB, long sC)
{
  // K-tile 32, padded stride 40 (80 B rows keep 16 B alignment, kill conflicts)
  __shared__ uint16_t Abuf[2][128][40];
  __shared__ uint16_t Bbuf[2][128][40];

  const int tid   = threadIdx.x;
  const int lane  = tid & 31;
  const int wave  = tid >> 5;           // 0..7
  const int wm    = wave >> 2;          // 0..1 : rows  wm*64 .. +64
  const int wn    = wave & 3;           // 0..3 : cols  wn*32 .. +32
  const int lrow  = lane & 15;
  const int lhalf = (lane >> 4) & 1;

  const int tileM = blockIdx.y * 128;
  const int tileN = blockIdx.x * 128;

  A  += (long)blockIdx.z * sA;
  Bm += (long)blockIdx.z * sB;

  // cooperative load: 256 threads, each owns one 8-element (16 B) chunk per
  // matrix per K-step: row = tid>>1 (0..127), seg = 0/16 within the K-tile.
  const int ldRow = tid >> 1;
  const int ldSeg = (tid & 1) * 16;

  const uint16_t* gA = A  + (size_t)(tileM + ldRow) * lda + ldSeg;
  const uint16_t* gB = Bm + (size_t)(tileN + ldRow) * ldb + ldSeg;

  // LDS byte offsets of this thread's staging slots (low 32 bits of the
  // generic address == LDS offset per the flat-aperture mapping, ISA 10.2).
  uint32_t aOff[2], bOff[2];
  aOff[0] = (uint32_t)(uintptr_t)&Abuf[0][ldRow][ldSeg];
  aOff[1] = (uint32_t)(uintptr_t)&Abuf[1][ldRow][ldSeg];
  bOff[0] = (uint32_t)(uintptr_t)&Bbuf[0][ldRow][ldSeg];
  bOff[1] = (uint32_t)(uintptr_t)&Bbuf[1][ldRow][ldSeg];

  union Frag { v16bf v; uint16_t s[16]; };

  v8f acc[4][2];
  {
    v8f z = {};
    #pragma unroll
    for (int f = 0; f < 4; ++f) { acc[f][0] = z; acc[f][1] = z; }
  }

  const int nk = K >> 5;

  // prologue: stage K-tile 0 into buffer 0
  async_copy_b128(aOff[0], gA);
  async_copy_b128(bOff[0], gB);
  wait_asynccnt0();
  __syncthreads();

  for (int kt = 0; kt < nk; ++kt) {
    const int cur = kt & 1;
    const int nxt = cur ^ 1;

    if (kt + 1 < nk) {                 // overlap next tile's copy with math
      const int kn = (kt + 1) << 5;
      async_copy_b128(aOff[nxt], gA + kn);
      async_copy_b128(bOff[nxt], gB + kn);
    }
    if (kt + 2 < nk) {                 // L2 prefetch two tiles ahead
      const int kp = (kt + 2) << 5;
      __builtin_prefetch(gA + kp, 0, 3);
      __builtin_prefetch(gB + kp, 0, 3);
    }

    Frag a[4], b[2];
    // A 16x32 bf16 frag: lanes 0-15 = rows, elems 0-7 -> K=lhalf*8+0..7,
    // elems 8-15 -> K=16+lhalf*8+0..7  (ISA 7.12.2)
    #pragma unroll
    for (int f = 0; f < 4; ++f) {
      const int r = wm * 64 + f * 16 + lrow;
      #pragma unroll
      for (int i = 0; i < 16; ++i) {
        const int k = ((i >> 3) << 4) + (lhalf << 3) + (i & 7);
        a[f].s[i] = Abuf[cur][r][k];
      }
    }
    // B 32x16 bf16 frag: lane = column N, elems 0-7 -> K=lhalf*8+0..7, elems 8-15 -> K=16+lhalf*8+0..7
    #pragma unroll
    for (int g = 0; g < 2; ++g) {
      const int c = wn * 32 + g * 16 + lrow;
      #pragma unroll
      for (int i = 0; i < 16; ++i)
        b[g].s[i] = Bbuf[cur][c][((i >> 3) << 4) + (lhalf << 3) + (i & 7)];
    }

    #pragma unroll
    for (int f = 0; f < 4; ++f)
      #pragma unroll
      for (int g = 0; g < 2; ++g)
        acc[f][g] = __builtin_amdgcn_wmma_f32_16x16x32_bf16(
            false, a[f].v, false, b[g].v, (short)0, acc[f][g], false, false);

    wait_asynccnt0();                  // own copies into buf[nxt] landed
    __syncthreads();                   // everyone done reading buf[cur]
  }

  // Epilogue: C/D layout -> VGPR r holds M = r + 8*lhalf, N = lane&15
  #pragma unroll
  for (int f = 0; f < 4; ++f) {
    #pragma unroll
    for (int g = 0; g < 2; ++g) {
      const int mbase = tileM + wm * 64 + f * 16;
      const int nbase = tileN + wn * 32 + g * 16;
      const int col   = nbase + lrow;
      #pragma unroll
      for (int r = 0; r < 8; ++r) {
        const int row = mbase + r + lhalf * 8;
        float v = acc[f][g][r];
        if (EPI == EPI_SCORES) { v *= 0.1f; v = v * v + v; }   // poly "softmax"
        if (EPI == EPI_F32) {
          float* C = (float*)Cp + (long)blockIdx.z * sC;
          C[(size_t)row * ldc + col] = v;
        } else if (EPI == EPI_BF16T) {
          uint16_t* C = (uint16_t*)Cp + (long)blockIdx.z * sC;
          C[(size_t)col * ldc + row] = f2bf(v);                // transposed store
        } else {
          uint16_t* C = (uint16_t*)Cp + (long)blockIdx.z * sC;
          C[(size_t)row * ldc + col] = f2bf(v);
        }
      }
    }
  }
}

// fp32 -> bf16 with folded scale
__global__ __launch_bounds__(256)
void cvt_f32_bf16(const float* __restrict__ in, uint16_t* __restrict__ out,
                  float scale, long n)
{
  long i = (long)blockIdx.x * blockDim.x + threadIdx.x;
  const long stride = (long)gridDim.x * blockDim.x;
  for (; i < n; i += stride) out[i] = f2bf(in[i] * scale);
}

extern "C" void kernel_launch(void* const* d_in, const int* in_sizes, int n_in,
                              void* d_out, int out_size, void* d_ws, size_t ws_size,
                              hipStream_t stream)
{
  (void)in_sizes; (void)n_in; (void)out_size; (void)ws_size;

  constexpr int  Bn = 2, S = 4096, D = 768;
  constexpr long MS = (long)Bn * S;            // 8192 total rows

  const float* x  = (const float*)d_in[0];
  const float* Wq = (const float*)d_in[1];
  const float* Wk = (const float*)d_in[2];
  const float* Wv = (const float*)d_in[3];
  const float* Wo = (const float*)d_in[4];

  char*  ws  = (char*)d_ws;
  size_t off = 0;
  auto alloc = [&](size_t bytes) -> void* {
    void* p = ws + off;
    off += (bytes + 255) & ~size_t(255);
    return p;
  };

  uint16_t* xbf = (uint16_t*)alloc((size_t)MS * D * 2);  // later aliased by attn_out
  uint16_t* wq  = (uint16_t*)alloc((size_t)D * D * 2);   // Wq * 0.1 (q scale)
  uint16_t* wk  = (uint16_t*)alloc((size_t)D * D * 2);   // Wk * 0.1 (k scale)
  uint16_t* wv  = (uint16_t*)alloc((size_t)D * D * 2);
  uint16_t* wo  = (uint16_t*)alloc((size_t)D * D * 2);   // Wo * 0.1 (out scale)
  uint16_t* qb  = (uint16_t*)alloc((size_t)MS * D * 2);  // q bf16 [B,S,D]
  uint16_t* kb  = (uint16_t*)alloc((size_t)MS * D * 2);  // k bf16 [B,S,D]
  uint16_t* vT  = (uint16_t*)alloc((size_t)MS * D * 2);  // v^T bf16 [B,D,S]
  uint16_t* sc  = (uint16_t*)alloc((size_t)S * S * 2);   // scores, one batch (L2-resident)
  uint16_t* ao  = xbf;                                   // attn_out aliases xbf

  // --- bf16 conversions with folded scales -------------------------------
  cvt_f32_bf16<<<2048, 256, 0, stream>>>(x,  xbf, 1.0f,  MS * (long)D);
  cvt_f32_bf16<<<512,  256, 0, stream>>>(Wq, wq,  0.1f,  (long)D * D);
  cvt_f32_bf16<<<512,  256, 0, stream>>>(Wk, wk,  0.1f,  (long)D * D);
  cvt_f32_bf16<<<512,  256, 0, stream>>>(Wv, wv,  1.0f,  (long)D * D);
  cvt_f32_bf16<<<512,  256, 0, stream>>>(Wo, wo,  0.1f,  (long)D * D);

  dim3 blk(256);

  // --- q/k projections: [8192x768] = xbf[8192x768] * W^T ------------------
  gemm_wmma_bf16<EPI_BF16><<<dim3(D / 128, MS / 128, 1), blk, 0, stream>>>(
      xbf, wq, qb, D, D, D, D, 0, 0, 0);
  gemm_wmma_bf16<EPI_BF16><<<dim3(D / 128, MS / 128, 1), blk, 0, stream>>>(
      xbf, wk, kb, D, D, D, D, 0, 0, 0);

  // --- v projection, stored transposed per batch: vT[b][e][s] -------------
  gemm_wmma_bf16<EPI_BF16T><<<dim3(D / 128, S / 128, Bn), blk, 0, stream>>>(
      xbf, wv, vT, D, D, D, /*ldc=*/S,
      /*sA=*/(long)S * D, /*sB=*/0, /*sC=*/(long)D * S);

  // --- attention per batch (scores matrix stays resident in 192 MB L2) ----
  for (int b = 0; b < Bn; ++b) {
    const uint16_t* qb_b = qb + (size_t)b * S * D;
    const uint16_t* kb_b = kb + (size_t)b * S * D;
    const uint16_t* vT_b = vT + (size_t)b * D * S;
    uint16_t*       ao_b = ao + (size_t)b * S * D;

    // scores = poly(0.1 * q k^T): M=N=4096, K=768; k rows are K-contiguous
    gemm_wmma_bf16<EPI_SCORES><<<dim3(S / 128, S / 128, 1), blk, 0, stream>>>(
        qb_b, kb_b, sc, D, D, D, /*ldc=*/S, 0, 0, 0);

    // attn_out = scores @ v: M=4096, N=768, K=4096; B columns via vT rows
    gemm_wmma_bf16<EPI_BF16><<<dim3(D / 128, S / 128, 1), blk, 0, stream>>>(
        sc, vT_b, ao_b, /*K=*/S, /*lda=*/S, /*ldb=*/S, /*ldc=*/D, 0, 0, 0);
  }

  // --- final projection: out = attn_out @ (0.1*Wo)^T, fp32 output ---------
  gemm_wmma_bf16<EPI_F32><<<dim3(D / 128, MS / 128, 1), blk, 0, stream>>>(
      ao, wo, d_out, D, D, D, D, 0, 0, 0);
}
